// TMK_18648747999750
// MI455X (gfx1250) — compile-verified, matrix operand order checked by reading, not run on hardware
//
#include <hip/hip_runtime.h>

typedef __attribute__((ext_vector_type(16))) __bf16        v16bf;
typedef __attribute__((ext_vector_type(8)))  float         v8f;
typedef __attribute__((ext_vector_type(4)))  float         v4f;
typedef __attribute__((ext_vector_type(4)))  unsigned int  v4u;

namespace {
constexpr int N_ROWS   = 16384;  // rows of x / out
constexpr int M_COLS   = 2048;   // design points == K-depth == out cols
constexpr int TN       = 128;    // output rows per workgroup
constexpr int TM       = 256;    // output cols per workgroup
constexpr int TK       = 32;     // bf16 WMMA K-depth per step
constexpr int NTHREADS = 256;    // 8 wave32s, arranged 2 (rows) x 4 (cols)

union Frag {
  v16bf v;
  v4u   q[2];   // two 16B halves: elements 0..7 and 8..15
};
}  // namespace

__global__ __launch_bounds__(NTHREADS)
void laplace_proj_wmma_bf16x3(const float* __restrict__ x,
                              const float* __restrict__ dp,
                              const float* __restrict__ B,   // chol_inv [M,M]
                              float* __restrict__ out)
{
  __shared__ v4f sDP[M_COLS];                      // 32 KB: all design points
  __shared__ __align__(16) __bf16 sAhi[TN * TK];   // 8 KB
  __shared__ __align__(16) __bf16 sAlo[TN * TK];   // 8 KB
  __shared__ __align__(16) __bf16 sBhi[TM * TK];   // 16 KB (transposed [col][k])
  __shared__ __align__(16) __bf16 sBlo[TM * TK];   // 16 KB

  const int t    = threadIdx.x;
  const int lane = t & 31;
  const int wave = t >> 5;
  const int wr   = wave >> 2;    // 0..1
  const int wc   = wave & 3;     // 0..3
  const int half = lane >> 4;    // lane group 0/1
  const int l16  = lane & 15;
  const int kb   = half * 8;     // per-lane K base inside a fragment

  const int n0 = blockIdx.x * TN;
  const int m0 = blockIdx.y * TM;

  // Preload every design point row (float4) into LDS once.
  for (int i = t; i < M_COLS; i += NTHREADS)
    sDP[i] = reinterpret_cast<const v4f*>(dp)[i];

  // A-tile generator: 2 threads per output row, 16 K-columns each.
  const int genRow = t >> 1;
  const int genK0  = (t & 1) * 16;
  const v4f xv = reinterpret_cast<const v4f*>(x)[n0 + genRow];

  v8f acc[4][4] = {};

  __syncthreads();

  for (int k0 = 0; k0 < M_COLS; k0 += TK) {
    // ---- stage A: K(n,k) = exp(-||x_n - dp_k||_1), split to bf16 hi/lo ----
    for (int kk = 0; kk < 16; ++kk) {
      const int kcol = genK0 + kk;
      const v4f d = sDP[k0 + kcol];
      const float l1 = __builtin_fabsf(xv.x - d.x) + __builtin_fabsf(xv.y - d.y)
                     + __builtin_fabsf(xv.z - d.z) + __builtin_fabsf(xv.w - d.w);
      const float kv = __expf(-l1);
      const __bf16 hi = (__bf16)kv;
      const __bf16 lo = (__bf16)(kv - (float)hi);
      sAhi[genRow * TK + kcol] = hi;
      sAlo[genRow * TK + kcol] = lo;
    }

    // ---- stage B: chol_inv[k0..k0+31, m0..m0+255] -> LDS transposed hi/lo ----
    const float* bp = B + (size_t)k0 * M_COLS + m0 + t;
    #pragma unroll
    for (int r = 0; r < TK; ++r) {
      const float v = bp[(size_t)r * M_COLS];
      const __bf16 hi = (__bf16)v;
      const __bf16 lo = (__bf16)(v - (float)hi);
      sBhi[t * TK + r] = hi;
      sBlo[t * TK + r] = lo;
    }
    if (k0 + TK < M_COLS)
      __builtin_prefetch(bp + (size_t)TK * M_COLS, 0, 1);  // next K-slab of B

    __syncthreads();

    // ---- load fragments and run 3-product bf16 WMMA (hi*hi + hi*lo + lo*hi) ----
    Frag ahi[4], alo[4];
    #pragma unroll
    for (int tn = 0; tn < 4; ++tn) {
      const int r = wr * 64 + tn * 16 + l16;
      const __bf16* ph = &sAhi[r * TK + kb];
      const __bf16* pl = &sAlo[r * TK + kb];
      ahi[tn].q[0] = *reinterpret_cast<const v4u*>(ph);
      ahi[tn].q[1] = *reinterpret_cast<const v4u*>(ph + 16);
      alo[tn].q[0] = *reinterpret_cast<const v4u*>(pl);
      alo[tn].q[1] = *reinterpret_cast<const v4u*>(pl + 16);
    }

    #pragma unroll
    for (int tm = 0; tm < 4; ++tm) {
      const int c = wc * 64 + tm * 16 + l16;
      const __bf16* ph = &sBhi[c * TK + kb];
      const __bf16* pl = &sBlo[c * TK + kb];
      Frag bhi, blo;
      bhi.q[0] = *reinterpret_cast<const v4u*>(ph);
      bhi.q[1] = *reinterpret_cast<const v4u*>(ph + 16);
      blo.q[0] = *reinterpret_cast<const v4u*>(pl);
      blo.q[1] = *reinterpret_cast<const v4u*>(pl + 16);
      #pragma unroll
      for (int tn = 0; tn < 4; ++tn) {
        acc[tn][tm] = __builtin_amdgcn_wmma_f32_16x16x32_bf16(
            false, ahi[tn].v, false, bhi.v, (short)0, acc[tn][tm], false, false);
        acc[tn][tm] = __builtin_amdgcn_wmma_f32_16x16x32_bf16(
            false, ahi[tn].v, false, blo.v, (short)0, acc[tn][tm], false, false);
        acc[tn][tm] = __builtin_amdgcn_wmma_f32_16x16x32_bf16(
            false, alo[tn].v, false, bhi.v, (short)0, acc[tn][tm], false, false);
      }
    }
    __syncthreads();
  }

  // ---- epilogue: f32 C/D layout (VGPR v -> row v + 8*half, col = lane&15) ----
  #pragma unroll
  for (int tn = 0; tn < 4; ++tn) {
    #pragma unroll
    for (int tm = 0; tm < 4; ++tm) {
      const int rbase = n0 + wr * 64 + tn * 16 + half * 8;
      const int cbase = m0 + wc * 64 + tm * 16 + l16;
      #pragma unroll
      for (int v = 0; v < 8; ++v)
        out[(size_t)(rbase + v) * M_COLS + cbase] = acc[tn][tm][v];
    }
  }
}

extern "C" void kernel_launch(void* const* d_in, const int* in_sizes, int n_in,
                              void* d_out, int out_size, void* d_ws, size_t ws_size,
                              hipStream_t stream) {
  (void)in_sizes; (void)n_in; (void)out_size; (void)d_ws; (void)ws_size;
  const float* x  = (const float*)d_in[0];   // [16384, 4]
  const float* dp = (const float*)d_in[1];   // [2048, 4]
  const float* ch = (const float*)d_in[2];   // [2048, 2048]
  float* out      = (float*)d_out;           // [16384, 2048]

  dim3 grid(N_ROWS / TN, M_COLS / TM);       // (128, 8)
  dim3 block(NTHREADS);
  laplace_proj_wmma_bf16x3<<<grid, block, 0, stream>>>(x, dp, ch, out);
}
